// HOMEModel_47519518163426
// MI455X (gfx1250) — matrix operand order repaired and checked by reading, not compiled, not run on hardware
//
#include <hip/hip_runtime.h>

typedef __attribute__((ext_vector_type(16))) _Float16 v16h;
typedef __attribute__((ext_vector_type(8)))  _Float16 v8h;
typedef __attribute__((ext_vector_type(8)))  float    v8f;

#define MAXTILES 40
#define V8F_ZERO {0.f,0.f,0.f,0.f,0.f,0.f,0.f,0.f}

// ---------------- WMMA helpers (CDNA5, wave32) ----------------
__device__ __forceinline__ v8f wmma_f16(v16h a, v16h b, v8f c) {
  return __builtin_amdgcn_wmma_f32_16x16x32_f16(false, a, false, b, (short)0, c, false, false);
}
__device__ __forceinline__ v16h mk16(v8h lo, v8h hi) {
  return __builtin_shufflevector(lo, hi, 0,1,2,3,4,5,6,7,8,9,10,11,12,13,14,15);
}

// ---------------- deterministic hash RNG ----------------
__device__ __forceinline__ unsigned hashu(unsigned x) {
  x ^= x >> 16; x *= 0x7feb352du; x ^= x >> 15; x *= 0x846ca68bu; x ^= x >> 16; return x;
}
__device__ __forceinline__ float u01(unsigned h) {
  return ((h >> 8) + 0.5f) * (1.0f / 16777216.0f);   // in (0,1)
}
__device__ __forceinline__ float rnorm(unsigned idx) {
  unsigned h1 = hashu(idx * 2654435761u + 0x9E3779B9u);
  unsigned h2 = hashu(h1 ^ (idx * 2246822519u + 0x85EBCA6Bu));
  return sqrtf(-2.0f * __logf(u01(h1))) * __cosf(6.28318530718f * u01(h2));
}
__device__ __forceinline__ float rgumbel(unsigned idx) {
  float u = u01(hashu(idx * 747796405u + 0xA511E9B3u));
  return -__logf(-__logf(u));
}
__device__ __forceinline__ unsigned ordenc(float f) {  // order-preserving float->u32
  unsigned u = __float_as_uint(f);
  return (u & 0x80000000u) ? ~u : (u | 0x80000000u);
}
__device__ __forceinline__ float sigmoidf(float x) { return 1.0f / (1.0f + __expf(-x)); }

// ---------------- weight pre-pack: f32 row-major (K x N) -> f16 K-major [Np][Kp], zero pad --
__global__ void k_pack(const float* __restrict__ W, int K, int N,
                       _Float16* __restrict__ Wt, int Kp, int Np, int nmat)
{
  int idx = blockIdx.x * blockDim.x + threadIdx.x;
  int per = Kp * Np;
  if (idx >= per * nmat) return;
  int m = idx / per, r = idx - m * per;
  int n = r / Kp, k = r - n * Kp;
  float v = (k < K && n < N) ? W[(size_t)m * K * N + (size_t)k * N + n] : 0.0f;
  Wt[idx] = (_Float16)v;
}

// ---------------- prep: types + eidx-group permutation + tile descriptors ----------------
__global__ void k_prep(const float* __restrict__ history, int* etype, int* eidx,
                       int* perm, int* tdesc, int* ntil)
{
  int b = threadIdx.x;
  if (b < 512) {
    int t = (int)history[(size_t)b * 4800 + 5];       // history[b,0,0,5]
    t = t < 0 ? 0 : (t > 9 ? 9 : t);
    const int table[10] = {0, 1, 4, 2, 3, 4, 4, 4, 4, 4};
    etype[b] = t;
    eidx[b]  = table[t];
  }
  __syncthreads();
  if (threadIdx.x == 0) {
    int c[5] = {0,0,0,0,0};
    for (int i = 0; i < 512; ++i) c[eidx[i]]++;
    int off[5]; int o = 0;
    for (int g = 0; g < 5; ++g) { off[g] = o; o += c[g]; }
    int cur[5]; for (int g = 0; g < 5; ++g) cur[g] = off[g];
    for (int i = 0; i < 512; ++i) { int g = eidx[i]; perm[cur[g]++] = i; }
    int nt = 0;
    for (int g = 0; g < 5; ++g)
      for (int r = 0; r < c[g]; r += 16) {
        tdesc[nt*3+0] = g;
        tdesc[nt*3+1] = off[g] + r;
        tdesc[nt*3+2] = (c[g] - r) < 16 ? (c[g] - r) : 16;
        ++nt;
      }
    *ntil = nt;   // <= 36 < MAXTILES
  }
}

__global__ void k_zero(float* h0, float* h1, unsigned* h0m, unsigned* h1m,
                       unsigned long long* best)
{
  int i = blockIdx.x * blockDim.x + threadIdx.x;
  if (i < 512 * 256) { h0[i] = 0.f; h1[i] = 0.f; }
  if (i < 512 * 128) { h0m[i] = 0u; h1m[i] = 0u; }   // f16 mirrors as u32 pairs
  if (i < 5 * 60 * 512) best[i] = 0ull;
}

// ---------------- feature extractor layer 1 (virtual A: ego||type-emb, K=22) ----------------
// out f16 (25600 x 256), relu.  16x64 per wave, single K chunk (Kp=32).
__global__ void k_fe1(const float* __restrict__ history, const float* __restrict__ emb,
                      const int* __restrict__ etype,
                      const _Float16* __restrict__ W1t,   // packed [256][32]
                      const float* __restrict__ b1,
                      _Float16* __restrict__ out)
{
  int wid = (blockIdx.x * blockDim.x + threadIdx.x) >> 5;
  if (wid >= 1600 * 4) return;
  int mt = wid >> 2, nt = wid & 3;
  int m0 = mt << 4, n0 = nt << 6;
  int l = threadIdx.x & 31;
  int row = m0 + (l & 15);
  int bb = row / 50, ss = row - bb * 50;
  const float* ego = history + (size_t)bb * 4800 + ss * 6;
  const float* te  = emb + etype[bb] * 16;
  int kbA = (l & 16) ? 8 : 0;
  int kbB = (l & 16) ? 8 : 0;
  v16h a;
#pragma unroll
  for (int j = 0; j < 16; ++j) {
    int k = kbA + (j < 8 ? j : j + 8);
    float v = 0.0f;
    if (k < 6) v = ego[k]; else if (k < 22) v = te[k - 6];
    a[j] = (_Float16)v;
  }
  const _Float16* bcol0 = W1t + (size_t)(n0 + (l & 15)) * 32 + kbB;
  v8f acc[4] = {V8F_ZERO, V8F_ZERO, V8F_ZERO, V8F_ZERO};
#pragma unroll
  for (int u = 0; u < 4; ++u) {
    const _Float16* bp = bcol0 + (size_t)(u * 16) * 32;
    v16h b = mk16(*(const v8h*)bp, *(const v8h*)(bp + 16));
    acc[u] = wmma_f16(a, b, acc[u]);
  }
  int rb = m0 + ((l & 16) ? 8 : 0);
#pragma unroll
  for (int u = 0; u < 4; ++u) {
    int col = n0 + u * 16 + (l & 15);
    float bv = b1[col];
#pragma unroll
    for (int e = 0; e < 8; ++e)
      out[(size_t)(rb + e) * 256 + col] = (_Float16)fmaxf(acc[u][e] + bv, 0.0f);
  }
}

// ------- generic WMMA GEMM: C = act(A_f16(MxK) @ Wt_f16 + bias); 16x64 per wave -------
// A: f16, lda (K multiple of 32). Wt packed [Np][Kp]. Output f32 or f16 (outHalf).
__global__ void k_gemm(const _Float16* __restrict__ A, int lda,
                       const _Float16* __restrict__ Wt, int Kp,
                       const float* __restrict__ bias,
                       void* __restrict__ Cout, int ldc,
                       int M, int N, int K, int act, int outHalf)
{
  int wid = (blockIdx.x * blockDim.x + threadIdx.x) >> 5;
  int ntm = M >> 4, ntn = (N + 63) >> 6;
  if (wid >= ntm * ntn) return;
  int mt = wid / ntn, nt = wid - mt * ntn;
  int m0 = mt << 4, n0 = nt << 6;
  int l = threadIdx.x & 31;
  int kbA = (l & 16) ? 8 : 0;
  int kbB = (l & 16) ? 8 : 0;
  const _Float16* arow  = A + (size_t)(m0 + (l & 15)) * lda;
  const _Float16* bcol0 = Wt + (size_t)(n0 + (l & 15)) * Kp;
  v8f acc[4] = {V8F_ZERO, V8F_ZERO, V8F_ZERO, V8F_ZERO};
  for (int k0 = 0; k0 < K; k0 += 32) {
    v16h a = mk16(*(const v8h*)(arow + k0 + kbA),
                  *(const v8h*)(arow + k0 + kbA + 16));
#pragma unroll
    for (int u = 0; u < 4; ++u) {
      const _Float16* bp = bcol0 + (size_t)(u * 16) * Kp + k0 + kbB;
      v16h b = mk16(*(const v8h*)bp, *(const v8h*)(bp + 16));
      acc[u] = wmma_f16(a, b, acc[u]);
    }
  }
  int rb = m0 + ((l & 16) ? 8 : 0);
#pragma unroll
  for (int u = 0; u < 4; ++u) {
    int col = n0 + u * 16 + (l & 15);
    if (col >= N) continue;
    float bv = bias ? bias[col] : 0.0f;
#pragma unroll
    for (int e = 0; e < 8; ++e) {
      float v = acc[u][e] + bv;
      if (act) v = fmaxf(v, 0.0f);
      if (outHalf) ((_Float16*)Cout)[(size_t)(rb + e) * ldc + col] = (_Float16)v;
      else         ((float*)Cout)[(size_t)(rb + e) * ldc + col] = v;
    }
  }
}

// ---------------- encoder GRU step GEMMs (grouped by eidx via perm), f16 operands ----------
// blockIdx.y==0: gi = X_row @ Wih[g,l] + bih ; blockIdx.y==1: gh = Hst @ Whh[g,l] + bhh
__global__ void k_enc_gemm(const _Float16* __restrict__ X, int xScale, int xOff,
                           const _Float16* __restrict__ Hst,
                           const _Float16* __restrict__ Wiht,   // packed 10x[768][256]
                           const _Float16* __restrict__ Whht,
                           const float* __restrict__ enc_bih, const float* __restrict__ enc_bhh,
                           const int* __restrict__ perm, const int* __restrict__ tdesc,
                           const int* __restrict__ ntil,
                           float* __restrict__ gi, float* __restrict__ gh, int layer)
{
  int wid = (blockIdx.x * blockDim.x + threadIdx.x) >> 5;
  int mt = wid / 12, nt = wid - mt * 12;
  if (mt >= *ntil) return;
  int g  = tdesc[mt*3+0];
  int rs = tdesc[mt*3+1];
  int nr = tdesc[mt*3+2];
  bool isGH = (blockIdx.y == 1);
  const _Float16* Wt = (isGH ? Whht : Wiht) + (size_t)(g * 2 + layer) * 768 * 256;
  const float*    bs = (isGH ? enc_bhh : enc_bih) + (size_t)(g * 2 + layer) * 768;
  float* Out = isGH ? gh : gi;
  int l = threadIdx.x & 31;
  int i = l & 15;
  int bi = perm[rs + (i < nr ? i : 0)];
  const _Float16* arow = isGH ? (Hst + (size_t)bi * 256)
                              : (X + (size_t)(bi * xScale + xOff) * 256);
  int n0 = nt << 6;
  int kbA = (l & 16) ? 8 : 0;
  int kbB = (l & 16) ? 8 : 0;
  const _Float16* bcol0 = Wt + (size_t)(n0 + (l & 15)) * 256;
  v8f acc[4] = {V8F_ZERO, V8F_ZERO, V8F_ZERO, V8F_ZERO};
  for (int k0 = 0; k0 < 256; k0 += 32) {
    v16h a = mk16(*(const v8h*)(arow + k0 + kbA),
                  *(const v8h*)(arow + k0 + kbA + 16));
#pragma unroll
    for (int u = 0; u < 4; ++u) {
      const _Float16* bp = bcol0 + (size_t)(u * 16) * 256 + k0 + kbB;
      v16h b = mk16(*(const v8h*)bp, *(const v8h*)(bp + 16));
      acc[u] = wmma_f16(a, b, acc[u]);
    }
  }
  int rb = (l & 16) ? 8 : 0;
#pragma unroll
  for (int u = 0; u < 4; ++u) {
    int col = n0 + u * 16 + (l & 15);
    float bv = bs[col];
#pragma unroll
    for (int e = 0; e < 8; ++e) {
      int r = rb + e;
      if (r < nr) {
        int bo = perm[rs + r];                 // scatter back to original batch order
        Out[(size_t)bo * 768 + col] = acc[u][e] + bv;
      }
    }
  }
}

// ---------------- GRU gate nonlinearity: h = (1-z)*n + z*h (+ f16 mirror) ----------------
__global__ void k_gate(const float* __restrict__ gi, const float* __restrict__ gh,
                       float* __restrict__ h, _Float16* __restrict__ hm)
{
  int idx = blockIdx.x * blockDim.x + threadIdx.x;   // 512*256
  if (idx >= 512 * 256) return;
  int b = idx >> 8, j = idx & 255;
  const float* gib = gi + (size_t)b * 768;
  const float* ghb = gh + (size_t)b * 768;
  float r = sigmoidf(gib[j]       + ghb[j]);
  float z = sigmoidf(gib[256 + j] + ghb[256 + j]);
  float n = tanhf(gib[512 + j] + r * ghb[512 + j]);
  float hn = (1.0f - z) * n + z * h[idx];
  h[idx]  = hn;
  hm[idx] = (_Float16)hn;
}

// ---------------- decoder first-step gi (K=2, trivially VALU) ----------------
__global__ void k_dec_gi0(const float* __restrict__ history,
                          const float* __restrict__ Wih0, const float* __restrict__ bih0,
                          float* __restrict__ gi)
{
  int idx = blockIdx.x * blockDim.x + threadIdx.x;   // 512*768
  if (idx >= 512 * 768) return;
  int b = idx / 768, n = idx - b * 768;
  float x0 = history[(size_t)b * 4800 + 49 * 6 + 0]; // ego[b,-1,0]
  float x1 = history[(size_t)b * 4800 + 49 * 6 + 1]; // ego[b,-1,1]
  gi[idx] = x0 * Wih0[n] + x1 * Wih0[768 + n] + bih0[n];
}

__global__ void k_confmean(const float* __restrict__ conf, float* __restrict__ confm)
{
  int b = blockIdx.x * blockDim.x + threadIdx.x;
  if (b < 512) {
    float s = 0.f;
    for (int t = 0; t < 60; ++t) s += conf[(size_t)b * 60 + t];
    confm[b] = s * (1.0f / 60.0f);
  }
}

// ------- heatmap layer1: relu((h1 + scale_t*noise) @ hp_W1 + b1) -> f16 (30720x256) -------
__global__ void k_th1(const float* __restrict__ h1,
                      const _Float16* __restrict__ Wt,   // packed [256][256]
                      const float* __restrict__ bias,
                      _Float16* __restrict__ out)
{
  int wid = (blockIdx.x * blockDim.x + threadIdx.x) >> 5;
  if (wid >= 1920 * 4) return;
  int mt = wid >> 2, nt = wid & 3;
  int m0 = mt << 4, n0 = nt << 6;
  int l = threadIdx.x & 31;
  int row = m0 + (l & 15);
  int t = row >> 9, bb = row & 511;                  // m = t*512 + b
  float scale = 0.1f * (float)t * (1.0f / 60.0f);
  const float* hr = h1 + (size_t)bb * 256;
  int kbA = (l & 16) ? 8 : 0;
  int kbB = (l & 16) ? 8 : 0;
  const _Float16* bcol0 = Wt + (size_t)(n0 + (l & 15)) * 256;
  v8f acc[4] = {V8F_ZERO, V8F_ZERO, V8F_ZERO, V8F_ZERO};
  for (int k0 = 0; k0 < 256; k0 += 32) {
    v16h a;
#pragma unroll
    for (int j = 0; j < 16; ++j) {
      int k = k0 + kbA + (j < 8 ? j : j + 8);
      a[j] = (_Float16)(hr[k] + scale * rnorm((unsigned)(row * 256 + k)));
    }
#pragma unroll
    for (int u = 0; u < 4; ++u) {
      const _Float16* bp = bcol0 + (size_t)(u * 16) * 256 + k0 + kbB;
      v16h b = mk16(*(const v8h*)bp, *(const v8h*)(bp + 16));
      acc[u] = wmma_f16(a, b, acc[u]);
    }
  }
  int rb = m0 + ((l & 16) ? 8 : 0);
#pragma unroll
  for (int u = 0; u < 4; ++u) {
    int col = n0 + u * 16 + (l & 15);
    float bv = bias[col];
#pragma unroll
    for (int e = 0; e < 8; ++e)
      out[(size_t)(rb + e) * 256 + col] = (_Float16)fmaxf(acc[u][e] + bv, 0.0f);
  }
}

// ------- heatmap layer2 fused with Gumbel-max categorical sampling (no 503MB logits) -------
// Block = 8 waves sharing one 64-col hp_W2 panel staged in LDS (32KB); waves take 8
// consecutive M-tiles. Cuts the dominant L2 weight stream 8x; B frags come from ds_load_b128.
__global__ void k_sample(const _Float16* __restrict__ A,    // f16 30720 x 256
                         const _Float16* __restrict__ W2t,  // packed [4096][256]
                         const float* __restrict__ b2,
                         unsigned long long* __restrict__ best) // [5][60][512]
{
  __shared__ v8h lw[2048];                       // 64 cols x 256 k, f16 (16B aligned)
  int nt  = blockIdx.x & 63;                     // column group (64 cols)
  int mtg = blockIdx.x >> 6;                     // 0..239
  int n0  = nt << 6;
  // ---- cooperative panel stage: 16384 halves = 2048 v8h, 8 per thread ----
  {
    const v8h* src = (const v8h*)(W2t + (size_t)n0 * 256);
    int tid = threadIdx.x;
#pragma unroll
    for (int i = 0; i < 8; ++i)
      lw[tid + i * 256] = src[tid + i * 256];
  }
  __syncthreads();
  int mt = mtg * 8 + (threadIdx.x >> 5);
  int m0 = mt << 4;
  int l = threadIdx.x & 31;
  const _Float16* arow = A + (size_t)(m0 + (l & 15)) * 256;
  __builtin_prefetch((const void*)arow, 0, 2);   // global_prefetch_b8
  int kbA = (l & 16) ? 8 : 0;
  int kbB = (l & 16) ? 8 : 0;
  const _Float16* lwh = (const _Float16*)lw;
  const _Float16* bcol0 = lwh + (size_t)(l & 15) * 256;
  v8f acc[4] = {V8F_ZERO, V8F_ZERO, V8F_ZERO, V8F_ZERO};
  for (int k0 = 0; k0 < 256; k0 += 32) {
    v16h a = mk16(*(const v8h*)(arow + k0 + kbA),
                  *(const v8h*)(arow + k0 + kbA + 16));
#pragma unroll
    for (int u = 0; u < 4; ++u) {
      const _Float16* bp = bcol0 + (size_t)(u * 16) * 256 + k0 + kbB;   // LDS
      v16h b = mk16(*(const v8h*)bp, *(const v8h*)(bp + 16));
      acc[u] = wmma_f16(a, b, acc[u]);
    }
  }
  int colv[4]; float bv[4];
#pragma unroll
  for (int u = 0; u < 4; ++u) { colv[u] = n0 + u * 16 + (l & 15); bv[u] = b2[colv[u]]; }
  int rhalf = (l & 16) ? 8 : 0;
  for (int e = 0; e < 8; ++e) {
    int m = m0 + rhalf + e;
    int t = m >> 9, bb = m & 511;
    for (int s = 0; s < 5; ++s) {
      unsigned sbase = (unsigned)((s * 60 + t) * 512 + bb) * 4096u;
      unsigned long long pk = 0ull;
#pragma unroll
      for (int u = 0; u < 4; ++u) {
        float v = acc[u][e] + bv[u] + rgumbel(sbase + (unsigned)colv[u]);
        unsigned long long cand = ((unsigned long long)ordenc(v) << 32) | (unsigned)colv[u];
        if (cand > pk) pk = cand;
      }
      // max-reduce across the 16 lanes of this half-wave (xor masks stay within half)
      for (int mk = 8; mk >= 1; mk >>= 1) {
        unsigned lo = __shfl_xor((unsigned)pk, mk, 32);
        unsigned hi = __shfl_xor((unsigned)(pk >> 32), mk, 32);
        unsigned long long o = ((unsigned long long)hi << 32) | lo;
        if (o > pk) pk = o;
      }
      if ((l & 15) == 0)
        atomicMax(&best[(size_t)(s * 60 + t) * 512 + bb], pk);  // deterministic argmax
    }
  }
}

// ---------------- final scatter: predictions (B,6,60,2) + confidences (B,6) ----------------
__global__ void k_output(const float* __restrict__ trajb,            // 512 x 120
                         const unsigned long long* __restrict__ best,
                         const float* __restrict__ confm,
                         float* __restrict__ out)
{
  int idx = blockIdx.x * blockDim.x + threadIdx.x;   // 512*60
  if (idx >= 512 * 60) return;
  int b = idx / 60, t = idx - b * 60;
  float* pb = out + (size_t)b * 6 * 60 * 2;
  pb[t * 2 + 0] = trajb[(size_t)b * 120 + t * 2 + 0];
  pb[t * 2 + 1] = trajb[(size_t)b * 120 + t * 2 + 1];
  const float CELL = 1.5625f, HALF = 0.78125f, GR0v = -50.0f;
  for (int s = 1; s < 6; ++s) {
    unsigned id = (unsigned)(best[(size_t)((s - 1) * 60 + t) * 512 + b] & 0xFFFFFFFFull);
    int xi = (int)(id & 63u), yi = (int)(id >> 6);
    pb[(s * 60 + t) * 2 + 0] = GR0v + xi * CELL + HALF;
    pb[(s * 60 + t) * 2 + 1] = GR0v + yi * CELL + HALF;
  }
  if (t == 0) {
    float cm = confm[b], d = 1.0f;
    for (int s = 0; s < 6; ++s) { out[368640 + b * 6 + s] = cm * d; d *= 0.9f; }
  }
}

// =======================================================================================
extern "C" void kernel_launch(void* const* d_in, const int* in_sizes, int n_in,
                              void* d_out, int out_size, void* d_ws, size_t ws_size,
                              hipStream_t stream)
{
  const float* history  = (const float*)d_in[0];
  const float* emb      = (const float*)d_in[1];
  const float* fe_W1    = (const float*)d_in[2];
  const float* fe_b1    = (const float*)d_in[3];
  const float* fe_W2    = (const float*)d_in[4];
  const float* fe_b2    = (const float*)d_in[5];
  const float* enc_Wih  = (const float*)d_in[6];
  const float* enc_Whh  = (const float*)d_in[7];
  const float* enc_bih  = (const float*)d_in[8];
  const float* enc_bhh  = (const float*)d_in[9];
  const float* dec_Wih0 = (const float*)d_in[10];
  const float* dec_Whh0 = (const float*)d_in[11];
  const float* dec_bih0 = (const float*)d_in[12];
  const float* dec_bhh0 = (const float*)d_in[13];
  const float* dec_Wih1 = (const float*)d_in[14];
  const float* dec_Whh1 = (const float*)d_in[15];
  const float* dec_bih1 = (const float*)d_in[16];
  const float* dec_bhh1 = (const float*)d_in[17];
  const float* hg_W1 = (const float*)d_in[18];
  const float* hg_b1 = (const float*)d_in[19];
  const float* hg_W2 = (const float*)d_in[20];
  const float* hg_b2 = (const float*)d_in[21];
  const float* hg_W3 = (const float*)d_in[22];
  const float* hg_b3 = (const float*)d_in[23];
  const float* ce_W1 = (const float*)d_in[24];
  const float* ce_b1 = (const float*)d_in[25];
  const float* ce_W2 = (const float*)d_in[26];
  const float* ce_b2 = (const float*)d_in[27];
  const float* hp_W1 = (const float*)d_in[28];
  const float* hp_b1 = (const float*)d_in[29];
  const float* hp_W2 = (const float*)d_in[30];
  const float* hp_b2 = (const float*)d_in[31];
  float* out = (float*)d_out;
  (void)in_sizes; (void)n_in; (void)out_size; (void)ws_size;

  char* base = (char*)d_ws;
  size_t off = 0;
  auto alloc = [&](size_t bytes) -> char* {
    off = (off + 255) & ~(size_t)255;
    char* p = base + off;
    off += bytes;
    return p;
  };
  // ints / small f32
  int* etype = (int*)alloc(512 * 4);
  int* eidx  = (int*)alloc(512 * 4);
  int* perm  = (int*)alloc(512 * 4);
  int* tdesc = (int*)alloc(MAXTILES * 3 * 4);
  int* ntil  = (int*)alloc(4);
  float* gi    = (float*)alloc((size_t)512 * 768 * 4);
  float* gh    = (float*)alloc((size_t)512 * 768 * 4);
  float* h0    = (float*)alloc((size_t)512 * 256 * 4);
  float* h1    = (float*)alloc((size_t)512 * 256 * 4);
  float* trajb = (float*)alloc((size_t)512 * 120 * 4);
  float* confb = (float*)alloc((size_t)512 * 60 * 4);
  float* confm = (float*)alloc((size_t)512 * 4);
  unsigned long long* best = (unsigned long long*)alloc((size_t)5 * 60 * 512 * 8);
  // packed f16 weights
  _Float16* feW1t   = (_Float16*)alloc((size_t)256 * 32 * 2);
  _Float16* feW2t   = (_Float16*)alloc((size_t)256 * 256 * 2);
  _Float16* encWiht = (_Float16*)alloc((size_t)10 * 768 * 256 * 2);
  _Float16* encWhht = (_Float16*)alloc((size_t)10 * 768 * 256 * 2);
  _Float16* decWhh0t= (_Float16*)alloc((size_t)768 * 256 * 2);
  _Float16* decWih1t= (_Float16*)alloc((size_t)768 * 256 * 2);
  _Float16* decWhh1t= (_Float16*)alloc((size_t)768 * 256 * 2);
  _Float16* hgW1t   = (_Float16*)alloc((size_t)256 * 256 * 2);
  _Float16* hgW2t   = (_Float16*)alloc((size_t)128 * 256 * 2);
  _Float16* hgW3t   = (_Float16*)alloc((size_t)128 * 128 * 2);
  _Float16* ceW1t   = (_Float16*)alloc((size_t)128 * 256 * 2);
  _Float16* ceW2t   = (_Float16*)alloc((size_t)64 * 128 * 2);
  _Float16* hpW1t   = (_Float16*)alloc((size_t)256 * 256 * 2);
  _Float16* hpW2t   = (_Float16*)alloc((size_t)4096 * 256 * 2);
  // f16 activations
  _Float16* h0f  = (_Float16*)alloc((size_t)512 * 256 * 2);
  _Float16* h1f  = (_Float16*)alloc((size_t)512 * 256 * 2);
  _Float16* hgt1 = (_Float16*)alloc((size_t)512 * 256 * 2);
  _Float16* hgt2 = (_Float16*)alloc((size_t)512 * 128 * 2);
  _Float16* cet1 = (_Float16*)alloc((size_t)512 * 128 * 2);
  _Float16* feat = (_Float16*)alloc((size_t)25600 * 256 * 2);
  _Float16* bigA = (_Float16*)alloc((size_t)30720 * 256 * 2);  // t1fe then th1 (aliased)
  _Float16* t1fe = bigA;
  _Float16* th1b = bigA;

  auto pack = [&](const float* W, int K, int N, _Float16* Wt, int Kp, int Np, int nmat) {
    int tot = Kp * Np * nmat;
    k_pack<<<(tot + 255) / 256, 256, 0, stream>>>(W, K, N, Wt, Kp, Np, nmat);
  };
  auto gemm = [&](const _Float16* A, int lda, const _Float16* Wt, int Kp, const float* bias,
                  void* C, int ldc, int M, int N, int K, int act, int outHalf) {
    int tiles = (M >> 4) * ((N + 63) >> 6);
    int blocks = (tiles + 7) / 8;
    k_gemm<<<blocks, 256, 0, stream>>>(A, lda, Wt, Kp, bias, C, ldc, M, N, K, act, outHalf);
  };

  // ---- one-time per-call weight packing (f32 -> f16 K-major, zero-padded) ----
  pack(fe_W1,   22,  256, feW1t,   32,  256, 1);
  pack(fe_W2,   256, 256, feW2t,   256, 256, 1);
  pack(enc_Wih, 256, 768, encWiht, 256, 768, 10);
  pack(enc_Whh, 256, 768, encWhht, 256, 768, 10);
  pack(dec_Whh0,256, 768, decWhh0t,256, 768, 1);
  pack(dec_Wih1,256, 768, decWih1t,256, 768, 1);
  pack(dec_Whh1,256, 768, decWhh1t,256, 768, 1);
  pack(hg_W1,   256, 256, hgW1t,   256, 256, 1);
  pack(hg_W2,   256, 128, hgW2t,   256, 128, 1);
  pack(hg_W3,   128, 120, hgW3t,   128, 128, 1);
  pack(ce_W1,   256, 128, ceW1t,   256, 128, 1);
  pack(ce_W2,   128, 60,  ceW2t,   128, 64,  1);
  pack(hp_W1,   256, 256, hpW1t,   256, 256, 1);
  pack(hp_W2,   256, 4096,hpW2t,   256, 4096,1);

  k_prep<<<1, 512, 0, stream>>>(history, etype, eidx, perm, tdesc, ntil);
  k_zero<<<600, 256, 0, stream>>>(h0, h1, (unsigned*)h0f, (unsigned*)h1f, best);

  // feature extractor
  k_fe1<<<800, 256, 0, stream>>>(history, emb, etype, feW1t, fe_b1, t1fe);
  gemm(t1fe, 256, feW2t, 256, fe_b2, feat, 256, 25600, 256, 256, 0, 1);

  // 2-layer GRU encoder, layers interleaved per timestep (equivalent to layered scans)
  for (int t = 0; t < 50; ++t) {
    k_enc_gemm<<<dim3(60, 2), 256, 0, stream>>>(feat, 50, t, h0f, encWiht, encWhht,
        enc_bih, enc_bhh, perm, tdesc, ntil, gi, gh, 0);
    k_gate<<<512, 256, 0, stream>>>(gi, gh, h0, h0f);
    k_enc_gemm<<<dim3(60, 2), 256, 0, stream>>>(h0f, 1, 0, h1f, encWiht, encWhht,
        enc_bih, enc_bhh, perm, tdesc, ntil, gi, gh, 1);
    k_gate<<<512, 256, 0, stream>>>(gi, gh, h1, h1f);
  }

  // decoder GRU cells (h0/h0f hold enc_h0, h1/h1f hold enc_h1)
  k_dec_gi0<<<1536, 256, 0, stream>>>(history, dec_Wih0, dec_bih0, gi);
  gemm(h0f, 256, decWhh0t, 256, dec_bhh0, gh, 768, 512, 768, 256, 0, 0);
  k_gate<<<512, 256, 0, stream>>>(gi, gh, h0, h0f);
  gemm(h0f, 256, decWih1t, 256, dec_bih1, gi, 768, 512, 768, 256, 0, 0);
  gemm(h1f, 256, decWhh1t, 256, dec_bhh1, gh, 768, 512, 768, 256, 0, 0);
  k_gate<<<512, 256, 0, stream>>>(gi, gh, h1, h1f);   // h1 = last_hidden

  // trajectory + confidence heads
  gemm(h1f,  256, hgW1t, 256, hg_b1, hgt1,  256, 512, 256, 256, 1, 1);
  gemm(hgt1, 256, hgW2t, 256, hg_b2, hgt2,  128, 512, 128, 256, 1, 1);
  gemm(hgt2, 128, hgW3t, 128, hg_b3, trajb, 120, 512, 120, 128, 0, 0);
  gemm(h1f,  256, ceW1t, 256, ce_b1, cet1,  128, 512, 128, 256, 1, 1);
  gemm(cet1, 128, ceW2t, 128, ce_b2, confb,  60, 512,  60, 128, 0, 0);
  k_confmean<<<2, 256, 0, stream>>>(confb, confm);

  // heatmap head + fused Gumbel-max sampling
  k_th1<<<960, 256, 0, stream>>>(h1, hpW1t, hp_b1, th1b);
  k_sample<<<15360, 256, 0, stream>>>(th1b, hpW2t, hp_b2, best);

  k_output<<<120, 256, 0, stream>>>(trajb, best, confm, out);
}
